// HALTransformerBlock_74844100100435
// MI455X (gfx1250) — matrix operand order repaired and checked by reading, not compiled, not run on hardware
//
#include <hip/hip_runtime.h>
#include <hip/hip_bf16.h>

typedef __attribute__((ext_vector_type(8)))  __bf16 v8bf;
typedef __attribute__((ext_vector_type(16))) __bf16 v16bf;
typedef __attribute__((ext_vector_type(8)))  float  v8f;
typedef __attribute__((ext_vector_type(4)))  int    v4i;

#define EPI_BF16 0
#define EPI_GELU_BF16 1
#define EPI_RES_F32 2

// ---- CDNA5 async global->LDS copy (ASYNCcnt) with portable fallback --------
#if defined(__has_builtin)
#if __has_builtin(__builtin_amdgcn_global_load_async_to_lds_b128)
#define USE_ASYNC_LDS 1
#endif
#endif
#ifndef USE_ASYNC_LDS
#define USE_ASYNC_LDS 0
#endif

typedef __attribute__((address_space(1))) v4i* gv4p;  // global int4*
typedef __attribute__((address_space(3))) v4i* lv4p;  // LDS int4*

__device__ __forceinline__ void async_copy16(void* lds_dst, const void* gsrc) {
#if USE_ASYNC_LDS
  __builtin_amdgcn_global_load_async_to_lds_b128((gv4p)gsrc, (lv4p)lds_dst, 0, 0);
#else
  *(v8bf*)lds_dst = *(const v8bf*)gsrc;
#endif
}
__device__ __forceinline__ void async_wait() {
#if USE_ASYNC_LDS
#if __has_builtin(__builtin_amdgcn_s_wait_asynccnt)
  __builtin_amdgcn_s_wait_asynccnt(0);
#else
  asm volatile("s_wait_asynccnt 0" ::: "memory");
#endif
#endif
}

// ---------------------------------------------------------------- cast f32->bf16 (Er)
__global__ __launch_bounds__(256) void cast_f32_bf16(const float* __restrict__ in,
                                                     __bf16* __restrict__ out, int n) {
  int i = blockIdx.x * blockDim.x + threadIdx.x;
  if (i < n) out[i] = (__bf16)in[i];
}

// ---------------------------------------------------------------- transpose+cast weights
// in: K x N (f32) -> out: N x K (bf16).  32x32 LDS tiles.
__global__ __launch_bounds__(256) void transpose_cast(const float* __restrict__ in,
                                                      __bf16* __restrict__ out, int K, int N) {
  __shared__ float tile[32][33];
  int k0 = blockIdx.y * 32, n0 = blockIdx.x * 32;
  int t = threadIdx.x, tr = t >> 5, tc = t & 31;
  #pragma unroll
  for (int i = 0; i < 4; i++)
    tile[tr + 8 * i][tc] = in[(size_t)(k0 + tr + 8 * i) * N + n0 + tc];
  __syncthreads();
  #pragma unroll
  for (int i = 0; i < 4; i++)
    out[(size_t)(n0 + tr + 8 * i) * K + k0 + tc] = (__bf16)tile[tc][tr + 8 * i];
}

// ---------------------------------------------------------------- transpose V: qkv -> vT[b,h,d,i]
__global__ __launch_bounds__(256) void transpose_v(const __bf16* __restrict__ qkv,
                                                   __bf16* __restrict__ vT) {
  __shared__ __bf16 tile[32][33];
  int i0 = blockIdx.x * 32, d0 = blockIdx.y * 32, bh = blockIdx.z;
  int b = bh >> 3, h = bh & 7;
  const __bf16* src = qkv + (size_t)(b * 1024) * 3072 + 2048 + h * 128;
  int t = threadIdx.x, tr = t >> 5, tc = t & 31;
  #pragma unroll
  for (int i = 0; i < 4; i++)
    tile[tr + 8 * i][tc] = src[(size_t)(i0 + tr + 8 * i) * 3072 + d0 + tc];
  __syncthreads();
  __bf16* dst = vT + (size_t)(bh * 128) * 1024;
  #pragma unroll
  for (int i = 0; i < 4; i++)
    dst[(size_t)(d0 + tr + 8 * i) * 1024 + i0 + tc] = tile[tc][tr + 8 * i];
}

// ---------------------------------------------------------------- LayerNorm (D=1024) -> bf16
__global__ __launch_bounds__(256) void ln_kernel(const float* __restrict__ x,
                                                 const float* __restrict__ g,
                                                 const float* __restrict__ bta,
                                                 __bf16* __restrict__ out, int D) {
  int row = blockIdx.x;
  const float* xr = x + (size_t)row * D;
  int t = threadIdx.x;
  float4 v = ((const float4*)xr)[t];
  float s  = v.x + v.y + v.z + v.w;
  float ss = v.x * v.x + v.y * v.y + v.z * v.z + v.w * v.w;
  for (int mk = 1; mk < 32; mk <<= 1) { s += __shfl_xor(s, mk); ss += __shfl_xor(ss, mk); }
  __shared__ float red[2][8];
  int w = t >> 5, ln = t & 31;
  if (ln == 0) { red[0][w] = s; red[1][w] = ss; }
  __syncthreads();
  if (w == 0) {
    float a  = (ln < 8) ? red[0][ln] : 0.f;
    float b2 = (ln < 8) ? red[1][ln] : 0.f;
    for (int mk = 1; mk < 8; mk <<= 1) { a += __shfl_xor(a, mk); b2 += __shfl_xor(b2, mk); }
    if (ln == 0) { red[0][0] = a; red[1][0] = b2; }
  }
  __syncthreads();
  float mu  = red[0][0] / D;
  float var = red[1][0] / D - mu * mu;
  float rs  = rsqrtf(var + 1e-5f);
  int c = t * 4;
  __bf16* o = out + (size_t)row * D + c;
  o[0] = (__bf16)((v.x - mu) * rs * g[c + 0] + bta[c + 0]);
  o[1] = (__bf16)((v.y - mu) * rs * g[c + 1] + bta[c + 1]);
  o[2] = (__bf16)((v.z - mu) * rs * g[c + 2] + bta[c + 2]);
  o[3] = (__bf16)((v.w - mu) * rs * g[c + 3] + bta[c + 3]);
}

// ---------------------------------------------------------------- tiled bf16 WMMA GEMM
// C(M,N) = A(M,K) @ Bt(N,K)^T + bias.  Block tile 128x256, wave tile 64x64,
// BK=64, double-buffered async LDS staging.  16 WMMAs per 8 fragment loads.
template <int EPI>
__global__ __launch_bounds__(256) void gemm_bf16(const __bf16* __restrict__ A,
                                                 const __bf16* __restrict__ Bt,  // N x K
                                                 const float* __restrict__ bias,
                                                 const float* __restrict__ resid,
                                                 void* __restrict__ out,
                                                 int M, int N, int K) {
  constexpr int LDT = 72;          // 144B rows: 16B-aligned chunks, conflict-free frag reads
  constexpr int ATILE = 128 * LDT; // elems per A LDS buffer
  constexpr int BTILE = 256 * LDT; // elems per B LDS buffer
  extern __shared__ __align__(16) char smem_g[];
  __bf16* const Asb = (__bf16*)smem_g;     // 2 x ATILE
  __bf16* const Bsb = Asb + 2 * ATILE;     // 2 x BTILE
  int tid = threadIdx.x, lane = tid & 31, w = tid >> 5;
  int wm = (w >> 2) * 64, wn = (w & 3) * 64;
  int m0 = blockIdx.y * 128, n0 = blockIdx.x * 256;
  int half = lane >> 4, l16 = lane & 15;

  auto load_tile = [&](__bf16* dst, const __bf16* src, int iters) {  // iters*32 rows x 64 bf16
    #pragma unroll
    for (int i = 0; i < iters; i++) {
      int c = tid + 256 * i;
      int row = c >> 3, cs = c & 7;
      async_copy16(&dst[row * LDT + cs * 8], src + (size_t)row * K + cs * 8);
    }
  };

  v8f acc[4][4];
  for (int i = 0; i < 4; i++)
    for (int j = 0; j < 4; j++)
      for (int e = 0; e < 8; e++) acc[i][j][e] = 0.f;

  const __bf16* Abase = A + (size_t)m0 * K;
  const __bf16* Bbase = Bt + (size_t)n0 * K;
  int KT = K >> 6;
  load_tile(Asb, Abase, 4);
  load_tile(Bsb, Bbase, 8);
  async_wait();
  __syncthreads();

  for (int kt = 0; kt < KT; kt++) {
    int cur = kt & 1, nxt = cur ^ 1;
    if (kt + 1 < KT) {
      int k0 = (kt + 1) << 6;
      load_tile(Asb + nxt * ATILE, Abase + k0, 4);
      load_tile(Bsb + nxt * BTILE, Bbase + k0, 8);
    }
    const __bf16* Ac = Asb + cur * ATILE;
    const __bf16* Bc = Bsb + cur * BTILE;
    #pragma unroll
    for (int kk = 0; kk < 64; kk += 32) {
      v16bf aF[4], bF[4];
      #pragma unroll
      for (int i = 0; i < 4; i++) {
        const __bf16* p = &Ac[(wm + 16 * i + l16) * LDT + kk];
        v8bf x0 = *(const v8bf*)(p + 16 * half);
        v8bf x1 = *(const v8bf*)(p + 16 * half + 8);
        for (int e = 0; e < 8; e++) { aF[i][e] = x0[e]; aF[i][8 + e] = x1[e]; }
      }
      #pragma unroll
      for (int j = 0; j < 4; j++) {
        const __bf16* p = &Bc[(wn + 16 * j + l16) * LDT + kk];
        v8bf x0 = *(const v8bf*)(p + 16 * half);
        v8bf x1 = *(const v8bf*)(p + 16 * half + 8);
        for (int e = 0; e < 8; e++) { bF[j][e] = x0[e]; bF[j][8 + e] = x1[e]; }
      }
      #pragma unroll
      for (int i = 0; i < 4; i++)
        #pragma unroll
        for (int j = 0; j < 4; j++)
          acc[i][j] = __builtin_amdgcn_wmma_f32_16x16x32_bf16(
              false, aF[i], false, bF[j], (short)0, acc[i][j], false, false);
    }
    async_wait();
    __syncthreads();
  }

  #pragma unroll
  for (int i = 0; i < 4; i++)
    #pragma unroll
    for (int j = 0; j < 4; j++) {
      int gn = n0 + wn + 16 * j + l16;
      float bv = bias[gn];
      #pragma unroll
      for (int r = 0; r < 8; r++) {
        int gm = m0 + wm + 16 * i + 8 * half + r;
        float v = acc[i][j][r] + bv;
        size_t idx = (size_t)gm * N + gn;
        if constexpr (EPI == EPI_GELU_BF16) {
          v = 0.5f * v * (1.0f + erff(v * 0.7071067811865475f));
          ((__bf16*)out)[idx] = (__bf16)v;
        } else if constexpr (EPI == EPI_BF16) {
          ((__bf16*)out)[idx] = (__bf16)v;
        } else {
          ((float*)out)[idx] = v + resid[idx];
        }
      }
    }
}

// ---------------------------------------------------------------- flash attention w/ relative pos
// grid (qtiles=8, nh=8, B=8), 256 threads. Srel[i][j] = q_i . Er[L-1-(i-j)] via band-WMMA + gather.
#define LK 136
__global__ __launch_bounds__(256) void attn_kernel(const __bf16* __restrict__ qkv,  // (B*L,3072)
                                                   const __bf16* __restrict__ vT,   // (B,8,128,1024)
                                                   const __bf16* __restrict__ Erb,  // (1024,128)
                                                   __bf16* __restrict__ y) {        // (B*L,1024)
  extern __shared__ __align__(16) char smem[];
  __bf16* Ks  = (__bf16*)smem;        // [128][LK] key-major
  __bf16* VTs = Ks + 128 * LK;        // [128][LK]: VT[dv][k]
  __bf16* Pw  = VTs + 128 * LK;       // per-wave [16][LK]
  const int L = 1024, HS = 128, QS = 3072;
  int b = blockIdx.z, h = blockIdx.y, qi0 = blockIdx.x * 128;
  int tid = threadIdx.x, lane = tid & 31, w = tid >> 5;
  int half = lane >> 4, l16 = lane & 15;
  int m0 = 16 * w;
  __bf16* Pme = Pw + w * 16 * LK;
  float* W1 = (float*)Pme;  // overlay scratch [16][40] f32

  const __bf16* qbase  = qkv + (size_t)(b * L) * QS + h * HS;
  const __bf16* kbase  = qbase + 1024;
  const __bf16* vtbase = vT + (size_t)((b * 8 + h) * 128) * 1024;

  v16bf aQ[4];  // wave's 16x128 Q strip in registers
  {
    int qrow = qi0 + m0 + l16;
    const __bf16* qp = qbase + (size_t)qrow * QS;
    #pragma unroll
    for (int d = 0; d < 4; d++) {
      v8bf c0 = *(const v8bf*)(qp + d * 32 + 16 * half);
      v8bf c1 = *(const v8bf*)(qp + d * 32 + 16 * half + 8);
      for (int e = 0; e < 8; e++) { aQ[d][e] = c0[e]; aQ[d][8 + e] = c1[e]; }
    }
  }

  float mrow[8], lrow[8];
  v8f o[8];
  for (int r = 0; r < 8; r++) { mrow[r] = -1e30f; lrow[r] = 0.f; }
  for (int f = 0; f < 8; f++)
    for (int e = 0; e < 8; e++) o[f][e] = 0.f;

  const float scale = 0.08838834764831845f;  // 1/sqrt(128)
  int nkt = qi0 / 128 + 1;

  for (int kt = 0; kt < nkt; kt++) {
    int kj0 = kt * 128;
    __syncthreads();
    #pragma unroll
    for (int i = 0; i < 8; i++) {  // K tile (key-major rows)
      int lin = (tid + i * 256) * 8, row = lin >> 7, col = lin & 127;
      async_copy16(&Ks[row * LK + col], kbase + (size_t)(kj0 + row) * QS + col);
    }
    #pragma unroll
    for (int i = 0; i < 8; i++) {  // VT tile (dv-major rows, pre-transposed)
      int lin = (tid + i * 256) * 8, row = lin >> 7, col = lin & 127;
      async_copy16(&VTs[row * LK + col], vtbase + (size_t)row * 1024 + kj0 + col);
    }
    async_wait();
    __syncthreads();

    v8f s[8];  // 16x128 score strip
    #pragma unroll
    for (int f = 0; f < 8; f++) {
      v8f a2; for (int e = 0; e < 8; e++) a2[e] = 0.f;
      #pragma unroll
      for (int d = 0; d < 4; d++) {
        v16bf bK;
        const __bf16* p = &Ks[(16 * f + l16) * LK + d * 32 + 16 * half];
        v8bf x0 = *(const v8bf*)p; v8bf x1 = *(const v8bf*)(p + 8);
        for (int e = 0; e < 8; e++) { bK[e] = x0[e]; bK[8 + e] = x1[e]; }
        a2 = __builtin_amdgcn_wmma_f32_16x16x32_bf16(false, aQ[d], false, bK, (short)0, a2, false, false);
      }
      s[f] = a2;
    }

    // Srel: band W1 = Qstrip @ Er[bs..bs+31]^T, then diagonal gather
    #pragma unroll 1
    for (int f = 0; f < 8; f++) {
      int bs = (L - 1) - (qi0 + m0) + (kj0 + 16 * f) - 15;
      #pragma unroll
      for (int tt = 0; tt < 2; tt++) {
        v8f a2; for (int e = 0; e < 8; e++) a2[e] = 0.f;
        int er = bs + 16 * tt + l16;
        bool ok = (er >= 0) && (er < L);
        const __bf16* ep = Erb + (size_t)(ok ? er : 0) * HS;
        #pragma unroll
        for (int d = 0; d < 4; d++) {
          v16bf bE;
          if (ok) {
            const __bf16* p = ep + d * 32 + 16 * half;
            v8bf x0 = *(const v8bf*)p; v8bf x1 = *(const v8bf*)(p + 8);
            for (int e = 0; e < 8; e++) { bE[e] = x0[e]; bE[8 + e] = x1[e]; }
          } else {
            for (int e = 0; e < 16; e++) bE[e] = (__bf16)0.f;
          }
          a2 = __builtin_amdgcn_wmma_f32_16x16x32_bf16(false, aQ[d], false, bE, (short)0, a2, false, false);
        }
        #pragma unroll
        for (int r = 0; r < 8; r++) W1[(8 * half + r) * 40 + tt * 16 + l16] = a2[r];
      }
      #pragma unroll
      for (int r = 0; r < 8; r++) {
        int mm = 8 * half + r;
        s[f][r] += W1[mm * 40 + (l16 - mm + 15)];
      }
    }

    // scale + causal mask
    int qb = qi0 + m0 + 8 * half;
    #pragma unroll
    for (int f = 0; f < 8; f++) {
      int kc = kj0 + 16 * f + l16;
      #pragma unroll
      for (int r = 0; r < 8; r++)
        s[f][r] = (kc <= qb + r) ? s[f][r] * scale : -1e30f;
    }
    // online softmax (rows live in 16-lane halves)
    float rmax[8];
    #pragma unroll
    for (int r = 0; r < 8; r++) {
      float v = s[0][r];
      for (int f = 1; f < 8; f++) v = fmaxf(v, s[f][r]);
      rmax[r] = v;
    }
    for (int mk = 1; mk < 16; mk <<= 1)
      #pragma unroll
      for (int r = 0; r < 8; r++) rmax[r] = fmaxf(rmax[r], __shfl_xor(rmax[r], mk));
    float alpha[8];
    #pragma unroll
    for (int r = 0; r < 8; r++) {
      float mn = fmaxf(mrow[r], rmax[r]);
      alpha[r] = __expf(mrow[r] - mn);
      mrow[r] = mn;
    }
    float rsum[8];
    for (int r = 0; r < 8; r++) rsum[r] = 0.f;
    #pragma unroll
    for (int f = 0; f < 8; f++)
      #pragma unroll
      for (int r = 0; r < 8; r++) {
        float p = __expf(s[f][r] - mrow[r]);
        s[f][r] = p;
        rsum[r] += p;
      }
    for (int mk = 1; mk < 16; mk <<= 1)
      #pragma unroll
      for (int r = 0; r < 8; r++) rsum[r] += __shfl_xor(rsum[r], mk);
    #pragma unroll
    for (int r = 0; r < 8; r++) lrow[r] = lrow[r] * alpha[r] + rsum[r];
    #pragma unroll
    for (int f = 0; f < 8; f++)
      for (int r = 0; r < 8; r++) o[f][r] *= alpha[r];

    // stage P (C-layout -> A-layout via wave-private LDS)
    #pragma unroll
    for (int f = 0; f < 8; f++)
      #pragma unroll
      for (int r = 0; r < 8; r++)
        Pme[(8 * half + r) * LK + 16 * f + l16] = (__bf16)s[f][r];
    v16bf aP[4];
    #pragma unroll
    for (int c = 0; c < 4; c++) {
      const __bf16* p = &Pme[l16 * LK + c * 32];
      v8bf x0 = *(const v8bf*)(p + 16 * half);
      v8bf x1 = *(const v8bf*)(p + 16 * half + 8);
      for (int e = 0; e < 8; e++) { aP[c][e] = x0[e]; aP[c][8 + e] = x1[e]; }
    }
    #pragma unroll
    for (int fv = 0; fv < 8; fv++) {
      #pragma unroll
      for (int c = 0; c < 4; c++) {
        v16bf bV;
        const __bf16* p = &VTs[(16 * fv + l16) * LK + c * 32 + 16 * half];
        v8bf x0 = *(const v8bf*)p; v8bf x1 = *(const v8bf*)(p + 8);
        for (int e = 0; e < 8; e++) { bV[e] = x0[e]; bV[8 + e] = x1[e]; }
        o[fv] = __builtin_amdgcn_wmma_f32_16x16x32_bf16(false, aP[c], false, bV, (short)0, o[fv], false, false);
      }
    }
  }

  float inv[8];
  for (int r = 0; r < 8; r++) inv[r] = 1.0f / lrow[r];
  #pragma unroll
  for (int f = 0; f < 8; f++)
    #pragma unroll
    for (int r = 0; r < 8; r++) {
      int row = qi0 + m0 + 8 * half + r;
      int col = h * HS + 16 * f + l16;
      y[(size_t)(b * L + row) * 1024 + col] = (__bf16)(o[f][r] * inv[r]);
    }
}

// ---------------------------------------------------------------- launch
extern "C" void kernel_launch(void* const* d_in, const int* in_sizes, int n_in,
                              void* d_out, int out_size, void* d_ws, size_t ws_size,
                              hipStream_t stream) {
  (void)in_sizes; (void)n_in; (void)out_size; (void)ws_size;
  const float* x     = (const float*)d_in[0];
  const float* Wqkv  = (const float*)d_in[1];
  const float* bqkv  = (const float*)d_in[2];
  const float* Wo    = (const float*)d_in[3];
  const float* bo    = (const float*)d_in[4];
  const float* Er    = (const float*)d_in[5];
  const float* g1    = (const float*)d_in[6];
  const float* b1    = (const float*)d_in[7];
  const float* g2    = (const float*)d_in[8];
  const float* b2    = (const float*)d_in[9];
  const float* Wfc   = (const float*)d_in[10];
  const float* bfc   = (const float*)d_in[11];
  const float* Wproj = (const float*)d_in[12];
  const float* bproj = (const float*)d_in[13];

  const int B = 8, L = 1024, D = 1024, F = 4096, BL = B * L;
  char* ws = (char*)d_ws;
  size_t off = 0;
  auto alloc = [&](size_t bytes) -> char* {
    char* p = ws + off;
    off += (bytes + 255) & ~(size_t)255;
    return p;
  };
  __bf16* WqkvT = (__bf16*)alloc((size_t)D * 3 * D * 2);   // (3D, D)
  __bf16* WoT   = (__bf16*)alloc((size_t)D * D * 2);       // (D, D)
  __bf16* WfcT  = (__bf16*)alloc((size_t)D * F * 2);       // (F, D)
  __bf16* WprojT= (__bf16*)alloc((size_t)F * D * 2);       // (D, F)
  __bf16* Er_b  = (__bf16*)alloc((size_t)L * 128 * 2);
  __bf16* xn    = (__bf16*)alloc((size_t)BL * D * 2);
  __bf16* qkvb  = (__bf16*)alloc((size_t)BL * 3 * D * 2);
  __bf16* vTb   = (__bf16*)alloc((size_t)BL * D * 2);
  __bf16* yb    = (__bf16*)alloc((size_t)BL * D * 2);
  float*  x1    = (float*) alloc((size_t)BL * D * 4);
  __bf16* x2n   = (__bf16*)alloc((size_t)BL * D * 2);
  __bf16* hb    = (__bf16*)alloc((size_t)BL * F * 2);

  transpose_cast<<<dim3(3 * D / 32, D / 32), 256, 0, stream>>>(Wqkv, WqkvT, D, 3 * D);
  transpose_cast<<<dim3(D / 32, D / 32),     256, 0, stream>>>(Wo,   WoT,   D, D);
  transpose_cast<<<dim3(F / 32, D / 32),     256, 0, stream>>>(Wfc,  WfcT,  D, F);
  transpose_cast<<<dim3(D / 32, F / 32),     256, 0, stream>>>(Wproj,WprojT,F, D);
  cast_f32_bf16<<<(L * 128 + 255) / 256, 256, 0, stream>>>(Er, Er_b, L * 128);

  const size_t gemm_lds = (size_t)(2 * 128 + 2 * 256) * 72 * sizeof(__bf16);  // 110592 B
  const size_t attn_lds = (size_t)3 * 128 * LK * sizeof(__bf16);              // 104448 B

  ln_kernel<<<BL, 256, 0, stream>>>(x, g1, b1, xn, D);
  gemm_bf16<EPI_BF16><<<dim3(3 * D / 256, BL / 128), 256, gemm_lds, stream>>>(
      xn, WqkvT, bqkv, nullptr, qkvb, BL, 3 * D, D);
  transpose_v<<<dim3(L / 32, 128 / 32, 64), 256, 0, stream>>>(qkvb, vTb);
  attn_kernel<<<dim3(8, 8, 8), 256, attn_lds, stream>>>(qkvb, vTb, Er_b, yb);
  gemm_bf16<EPI_RES_F32><<<dim3(D / 256, BL / 128), 256, gemm_lds, stream>>>(
      yb, WoT, bo, x, x1, BL, D, D);
  ln_kernel<<<BL, 256, 0, stream>>>(x1, g2, b2, x2n, D);
  gemm_bf16<EPI_GELU_BF16><<<dim3(F / 256, BL / 128), 256, gemm_lds, stream>>>(
      x2n, WfcT, bfc, nullptr, hb, BL, F, D);
  gemm_bf16<EPI_RES_F32><<<dim3(D / 256, BL / 128), 256, gemm_lds, stream>>>(
      hb, WprojT, bproj, x1, (float*)d_out, BL, D, F);
}
